// LocalFlowSmoothLoss_20109036880112
// MI455X (gfx1250) — matrix operand order repaired and checked by reading, not compiled, not run on hardware
//
#include <hip/hip_runtime.h>
#include <math.h>

// ---------------------------------------------------------------------------
// LocalFlowSmoothLoss for MI455X (gfx1250, wave32, WMMA)
//
// Reduction identity: sum(conv(dif, w)) == sum_p dif(p) * W_eff(p) where
// W_eff depends only on clamped distances to each boundary (w is symmetric
// per dimension). box = separable 7-tap sums: E via WMMA banded matmul,
// A via rolling sums through a zero-padded LDS plane, R via rolling window
// in the reduction kernel.
// ---------------------------------------------------------------------------

typedef __attribute__((ext_vector_type(16))) _Float16 v16h;
typedef __attribute__((ext_vector_type(8)))  float    v8f;

#define BDIM 4
#define RDIM 256
#define ADIM 256
#define EDIM 64
#define AE   (ADIM * EDIM)                     // 16384
#define RAE  ((size_t)RDIM * AE)               // 4194304
#define SAE_FLOATS ((size_t)BDIM * 3 * RAE)    // 50331648 floats (~201 MB)
#define INV343 (1.0f / 343.0f)
#define PROWS (ADIM + 6)                       // padded rows: -3..258
#define PLANE_BYTES (PROWS * EDIM * 2)         // 33536 B f16

// ---------------------------------------------------------------------------
// Kernel 0: Gaussian softmax weights -> boundary-weight tables, zero accum.
// ---------------------------------------------------------------------------
__global__ __launch_bounds__(1024) void k_init(float* __restrict__ W3,
                                               float* __restrict__ WE,
                                               double* __restrict__ acc) {
  __shared__ float w[343];
  __shared__ float w3s[784];
  __shared__ float ssum;
  const int t = threadIdx.x;
  if (t < 343) {
    int i = t / 49, j = (t / 7) % 7, k = t % 7;
    float dx = (float)(i - 3), dy = (float)(j - 3), dz = (float)(k - 3);
    float d = sqrtf(dx * dx + dy * dy + dz * dz);
    w[t] = __expf(__expf(-d * 1.25f));   // exp(softmax-logit), logit = exp(-d/0.8)
  }
  __syncthreads();
  if (t == 0) {
    float s = 0.f;
    for (int k = 0; k < 343; ++k) s += w[k];
    ssum = s;
  }
  __syncthreads();
  if (t < 343) w[t] = w[t] / ssum;
  __syncthreads();
  if (t < 784) {                         // t = (i*7 + j)*16 + eidx
    int i = t / 112, j = (t / 16) % 7, eidx = t % 16;
    int le = eidx >> 2, he = eidx & 3;
    float s = 0.f;
    for (int k = 3 - le; k <= 3 + he; ++k) s += w[i * 49 + j * 7 + k];
    w3s[t] = s;
    W3[t] = s;
  }
  __syncthreads();
  if (t < 16) {
    float s = 0.f;
    for (int ij = 0; ij < 49; ++ij) s += w3s[ij * 16 + t];
    WE[t] = s;
  }
  if (t < 2) acc[t] = 0.0;
}

// ---------------------------------------------------------------------------
// Kernel 1: per-(b,r) plane. flow1 = flow*bin*obj into zero-padded LDS plane
// (f16), rolling 7-tap sums along A (registers, branch-free thanks to the
// padded halo), write back in place, then 7-tap along E as a banded matmul
// D(16x64) = SA(16x64) x Band(64x64) via V_WMMA_F32_16X16X32_F16.
// Thread t: e = t&63, owns rows a = (t>>6)*64 .. +63 at that e-column.
// ---------------------------------------------------------------------------
__global__ __launch_bounds__(256) void k_plane(const float* __restrict__ pcl1,
                                               const float* __restrict__ flow,
                                               float* __restrict__ sae) {
  extern __shared__ __align__(16) char smem[];
  _Float16* plane = (_Float16*)smem;     // [PROWS][64]; row a lives at (a+3)*64

  const int t = threadIdx.x;
  const int e = t & 63;
  const int a0 = (t >> 6) * 64;
  const int L = t & 31;                  // lane within wave32
  const int wv = t >> 5;                 // wave id (0..7)
  const int b = blockIdx.x >> 8;
  const int r = blockIdx.x & 255;

  // zero the halo rows (-3..-1 and 256..258) once; covered by first barrier
  if (t < 192) {
    plane[t] = (_Float16)0.f;
    plane[(ADIM + 3) * 64 + t] = (_Float16)0.f;
  }

  const size_t baseP = (size_t)b * 2 * RAE + (size_t)r * AE;
  const size_t baseF = (size_t)b * 3 * RAE + (size_t)r * AE;

  // obj for this thread's 64 rows stays in registers (reused for 3 channels)
  float obj[64];
#pragma unroll
  for (int i = 0; i < 64; ++i) {
    size_t idx = baseP + (size_t)(a0 + i) * 64 + e;
    float p0 = pcl1[idx];
    float p1 = pcl1[idx + RAE];
    obj[i] = 1.0f / (1.0f + __expf(p0 - p1));   // softmax class-1 = sigmoid(p1-p0)
  }

  for (int c = 0; c < 3; ++c) {
    const float bin = (c == 0) ? 0.2f : 0.0175f;

    // flow1 into the padded plane
#pragma unroll
    for (int i = 0; i < 64; ++i) {
      float f = flow[baseF + (size_t)c * RAE + (size_t)(a0 + i) * 64 + e];
      plane[(a0 + i + 3) * 64 + e] = (_Float16)(f * bin * obj[i]);
    }
    __syncthreads();

    // rolling 7-tap along A (halo rows are zero -> no branches)
    float sa[64];
    {
      float s = 0.f;
#pragma unroll
      for (int da = -3; da <= 3; ++da) s += (float)plane[(a0 + da + 3) * 64 + e];
      sa[0] = s;
#pragma unroll
      for (int i = 1; i < 64; ++i) {
        s += (float)plane[(a0 + i + 6) * 64 + e]     // add row a+3
           - (float)plane[(a0 + i - 1) * 64 + e];    // sub row a-4
        sa[i] = s;
      }
    }
    __syncthreads();
#pragma unroll
    for (int i = 0; i < 64; ++i) plane[(a0 + i + 3) * 64 + e] = (_Float16)sa[i];
    __syncthreads();

    // 7-tap along E as banded matmul; wave handles rows [wv*32, wv*32+32)
    const size_t outbase = (size_t)(b * 3 + c) * RAE + (size_t)r * AE;
    const int rowA = wv * 32 + (L & 15);           // M = L%16 of batch 0
    for (int nt = 0; nt < 4; ++nt) {
      const int n = nt * 16 + (L & 15);            // output E index for this lane
      v8f acc0 = {}, acc1 = {};
#pragma unroll
      for (int kb = 0; kb < 2; ++kb) {
        // band B fragment (32x16 f16): lanes0-15 K=0..15, lanes16-31 K=16..31
        v16h bf;
#pragma unroll
        for (int rr = 0; rr < 8; ++rr) {
          int Kb = kb * 32 + ((L & 16) ? 16 : 0) + 2 * rr;
          int d0 = Kb - n, d1 = Kb + 1 - n;
          bf[2 * rr]     = (_Float16)((d0 >= -3 && d0 <= 3) ? 1.0f : 0.0f);
          bf[2 * rr + 1] = (_Float16)((d1 >= -3 && d1 <= 3) ? 1.0f : 0.0f);
        }
        // A fragments: lane's 16 f16 = two contiguous 16B runs -> ds_load_b128
        const int kbase = kb * 32 + ((L & 16) ? 16 : 0);
        union { v16h h; uint4 q[2]; } A0, A1;
        A0.q[0] = *(const uint4*)&plane[(rowA + 3) * 64 + kbase];
        A0.q[1] = *(const uint4*)&plane[(rowA + 3) * 64 + kbase + 8];
        A1.q[0] = *(const uint4*)&plane[(rowA + 19) * 64 + kbase];
        A1.q[1] = *(const uint4*)&plane[(rowA + 19) * 64 + kbase + 8];
        acc0 = __builtin_amdgcn_wmma_f32_16x16x32_f16(
            false, A0.h, false, bf, (short)0, acc0, false, false);
        acc1 = __builtin_amdgcn_wmma_f32_16x16x32_f16(
            false, A1.h, false, bf, (short)0, acc1, false, false);
      }
      // C/D layout: lanes0-15 rows M=v, lanes16-31 rows M=8+v; N = L%16
#pragma unroll
      for (int v = 0; v < 8; ++v) {
        int M = v + ((L & 16) ? 8 : 0);
        sae[outbase + (size_t)(wv * 32 + M) * 64 + n]      = acc0[v];
        sae[outbase + (size_t)(wv * 32 + 16 + M) * 64 + n] = acc1[v];
      }
    }
    __syncthreads();
  }
}

// ---------------------------------------------------------------------------
// Kernel 2: R-direction 7-tap as rolling window over an 8-row strip,
// recompute obj/flow1, dif = (flow1 - SRAE/343)^2, weight by W_eff,
// block-reduce in f64, atomic accumulate.
// ---------------------------------------------------------------------------
__global__ __launch_bounds__(256) void k_reduce(const float* __restrict__ pcl1,
                                                const float* __restrict__ flow,
                                                const float* __restrict__ sae,
                                                const float* __restrict__ W3,
                                                const float* __restrict__ WE,
                                                double* __restrict__ acc) {
  __shared__ double redN[256];
  __shared__ double redD[256];

  const int t = threadIdx.x;
  const size_t flat = (size_t)blockIdx.x * 256 + t;
  const int e  = (int)(flat & 63);
  const int a  = (int)((flat >> 6) & 255);
  const int rc = (int)((flat >> 14) & 31);
  const int b  = (int)(flat >> 19);
  const int r0 = rc * 8;

  size_t sb[3];
  float win[3];
#pragma unroll
  for (int c = 0; c < 3; ++c) {
    sb[c] = (size_t)(b * 3 + c) * RAE + (size_t)a * 64 + e;
    win[c] = 0.f;
  }
  for (int rr = r0 - 3; rr <= r0 + 3; ++rr) {
    if (rr >= 0 && rr < RDIM) {
#pragma unroll
      for (int c = 0; c < 3; ++c) win[c] += sae[sb[c] + (size_t)rr * AE];
    }
  }
  if (r0 + 8 < RDIM) {                       // global_prefetch_b8 of next strip
#pragma unroll
    for (int c = 0; c < 3; ++c)
      __builtin_prefetch(&sae[sb[c] + (size_t)(r0 + 8) * AE], 0, 1);
  }

  const int eidx = ((e < 3 ? e : 3) << 2) | ((63 - e) < 3 ? (63 - e) : 3);
  const float WEe = WE[eidx];
  const float bins[3] = {0.2f, 0.0175f, 0.0175f};
  const size_t pb = (size_t)b * 2 * RAE + (size_t)a * 64 + e;
  const size_t fb = (size_t)b * 3 * RAE + (size_t)a * 64 + e;

  double numd = 0.0, dend = 0.0;
  for (int j = 0; j < 8; ++j) {
    const int r = r0 + j;
    const size_t ro = (size_t)r * AE;
    float p0 = pcl1[pb + ro];
    float p1 = pcl1[pb + RAE + ro];
    float obj = 1.0f / (1.0f + __expf(p0 - p1));
    dend += (double)obj;

    float s = 0.f;
#pragma unroll
    for (int c = 0; c < 3; ++c) {
      float fl = flow[fb + (size_t)c * RAE + ro] * bins[c] * obj;
      float d = fl - win[c] * INV343;
      s += d * d;
    }

    float wgt;
    if (r >= 3 && r <= RDIM - 4 && a >= 3 && a <= ADIM - 4) {
      wgt = WEe;                                   // interior in R and A
    } else {
      int lr = r < 3 ? r : 3, hr = (RDIM - 1 - r) < 3 ? (RDIM - 1 - r) : 3;
      int la = a < 3 ? a : 3, ha = (ADIM - 1 - a) < 3 ? (ADIM - 1 - a) : 3;
      wgt = 0.f;
      for (int i = 3 - lr; i <= 3 + hr; ++i)
        for (int j2 = 3 - la; j2 <= 3 + ha; ++j2)
          wgt += W3[(i * 7 + j2) * 16 + eidx];
    }
    numd += (double)(wgt * s);

    int radd = r + 4, rsub = r - 3;
    if (radd < RDIM) {
#pragma unroll
      for (int c = 0; c < 3; ++c) win[c] += sae[sb[c] + (size_t)radd * AE];
    }
    if (rsub >= 0) {
#pragma unroll
      for (int c = 0; c < 3; ++c) win[c] -= sae[sb[c] + (size_t)rsub * AE];
    }
  }

  redN[t] = numd;
  redD[t] = dend;
  __syncthreads();
  for (int off = 128; off > 0; off >>= 1) {
    if (t < off) {
      redN[t] += redN[t + off];
      redD[t] += redD[t + off];
    }
    __syncthreads();
  }
  if (t == 0) {
    atomicAdd(&acc[0], redN[0]);
    atomicAdd(&acc[1], redD[0]);
  }
}

__global__ void k_final(const double* __restrict__ acc, float* __restrict__ out) {
  out[0] = (float)(acc[0] / acc[1]);
}

// ---------------------------------------------------------------------------
extern "C" void kernel_launch(void* const* d_in, const int* in_sizes, int n_in,
                              void* d_out, int out_size, void* d_ws, size_t ws_size,
                              hipStream_t stream) {
  (void)in_sizes; (void)n_in; (void)out_size; (void)ws_size;
  const float* pcl1 = (const float*)d_in[0];
  const float* flow = (const float*)d_in[1];
  // d_in[2] (coords) is unused by the reference forward.
  float* out = (float*)d_out;

  // workspace layout: SAE (201.3 MB) | W3 (784 f) | WE (16 f) | acc (2 f64)
  float*  sae = (float*)d_ws;
  float*  W3  = sae + SAE_FLOATS;
  float*  WE  = W3 + 784;
  double* acc = (double*)(WE + 16);

  k_init<<<1, 1024, 0, stream>>>(W3, WE, acc);
  k_plane<<<BDIM * RDIM, 256, PLANE_BYTES, stream>>>(pcl1, flow, sae);
  k_reduce<<<(BDIM * 32 * ADIM * EDIM) / 256, 256, 0, stream>>>(pcl1, flow, sae, W3, WE, acc);
  k_final<<<1, 1, 0, stream>>>(acc, out);
}
